// NeoX20BModel_42460046688648
// MI455X (gfx1250) — matrix operand order repaired and checked by reading, not compiled, not run on hardware
//
#include <hip/hip_runtime.h>
#include <hip/hip_bf16.h>

// ---------------- model constants (hardcoded from reference) ----------------
constexpr int S_  = 2048;
constexpr int H_  = 2048;
constexpr int NH_ = 16;
constexpr int HD_ = 128;   // H/NH
constexpr int L_  = 2;
constexpr int V_  = 32000;
constexpr int ROT_ = 32;   // HD*0.25
constexpr int H3_ = 3 * H_;   // 6144
constexpr int H4_ = 4 * H_;   // 8192
constexpr float EPS_ = 1e-5f;

// ---------------- WMMA types ----------------
typedef __attribute__((ext_vector_type(16))) _Float16 v16h;
typedef __attribute__((ext_vector_type(8)))  float    v8f;

struct Pack16 { unsigned u[8]; };   // 32 bytes == 16 halves
static __device__ __forceinline__ v16h as_v16h(const Pack16& p) {
    return __builtin_bit_cast(v16h, p);
}
static __device__ __forceinline__ v8f wmma_f16(v16h a, v16h b, v8f c) {
    return __builtin_amdgcn_wmma_f32_16x16x32_f16(false, a, false, b, (short)0, c, false, false);
}

// ---------------- async global->LDS copy (gfx1250 ASYNCcnt path), guarded ----------------
#if __has_builtin(__builtin_amdgcn_global_load_async_to_lds_b128)
#define HAVE_ASYNC_LDS 1
typedef __attribute__((__vector_size__(4 * sizeof(int)))) int v4i_;
#else
#define HAVE_ASYNC_LDS 0
#endif

static __device__ __forceinline__ void copy16B_to_lds(const void* g, void* l) {
#if HAVE_ASYNC_LDS
    __builtin_amdgcn_global_load_async_to_lds_b128((v4i_*)g, (v4i_*)l, 0, 0);
#else
    *(uint4*)l = *(const uint4*)g;
#endif
}
static __device__ __forceinline__ void wait_async_prev() {   // allow 2 in flight (next tile)
#if HAVE_ASYNC_LDS
    asm volatile("s_wait_asynccnt 0x2" ::: "memory");
#endif
}
static __device__ __forceinline__ void wait_async_all() {
#if HAVE_ASYNC_LDS
    asm volatile("s_wait_asynccnt 0x0" ::: "memory");
#endif
}

// ---------------- fp32 -> f16 weight conversion ----------------
__global__ __launch_bounds__(256) void cvt_f32_to_f16(const float* __restrict__ src,
                                                      _Float16* __restrict__ dst,
                                                      size_t n) {
    size_t stride = (size_t)gridDim.x * blockDim.x;
    for (size_t i = (size_t)blockIdx.x * blockDim.x + threadIdx.x; i < n; i += stride)
        dst[i] = (_Float16)src[i];
}

// ---------------- embedding gather ----------------
__global__ __launch_bounds__(256) void embed_kernel(const int* __restrict__ tok,
                                                    const float* __restrict__ emb,
                                                    float* __restrict__ h) {
    int row = blockIdx.x;
    int t = tok[row];
    const float* src = emb + (size_t)t * H_;
    float* dst = h + (size_t)row * H_;
    for (int i = threadIdx.x; i < H_; i += 256) dst[i] = src[i];
}

// ---------------- layernorm (fp32 in, f16 out) ----------------
__global__ __launch_bounds__(256) void layernorm_kernel(const float* __restrict__ x,
                                                        const float* __restrict__ g,
                                                        const float* __restrict__ b,
                                                        _Float16* __restrict__ out) {
    __shared__ float sbuf[256];
    int row = blockIdx.x;
    int tid = threadIdx.x;
    const float* xr = x + (size_t)row * H_;
    float s = 0.f, ss = 0.f;
    for (int i = tid; i < H_; i += 256) { float v = xr[i]; s += v; ss += v * v; }
    sbuf[tid] = s; __syncthreads();
    for (int off = 128; off > 0; off >>= 1) { if (tid < off) sbuf[tid] += sbuf[tid + off]; __syncthreads(); }
    float mean = sbuf[0] / (float)H_;
    __syncthreads();
    sbuf[tid] = ss; __syncthreads();
    for (int off = 128; off > 0; off >>= 1) { if (tid < off) sbuf[tid] += sbuf[tid + off]; __syncthreads(); }
    float var = sbuf[0] / (float)H_ - mean * mean;
    float rstd = rsqrtf(var + EPS_);
    _Float16* orow = out + (size_t)row * H_;
    for (int i = tid; i < H_; i += 256)
        orow[i] = (_Float16)((xr[i] - mean) * rstd * g[i] + b[i]);
}

// ---------------- RoPE on q and k inside qkv buffer (f16 in-place) ----------------
__global__ __launch_bounds__(256) void rope_kernel(_Float16* __restrict__ qkv) {
    int idx = blockIdx.x * 256 + threadIdx.x;       // S*NH*16 threads
    if (idx >= S_ * NH_ * (ROT_ / 2)) return;
    int d = idx & 15;
    int head = (idx >> 4) % NH_;
    int s = idx / (16 * NH_);
    float inv = expf(-((float)(2 * d) / (float)ROT_) * 9.210340371976184f); // ln(10000)
    float ang = (float)s * inv;
    float c = cosf(ang), sn = sinf(ang);
    size_t base = (size_t)s * H3_ + (size_t)head * (3 * HD_);
    #pragma unroll
    for (int w = 0; w < 2; ++w) {                   // w=0: q, w=1: k
        size_t p = base + (size_t)w * HD_;
        float x0 = (float)qkv[p + d];
        float x1 = (float)qkv[p + d + 16];
        qkv[p + d]      = (_Float16)(x0 * c - x1 * sn);
        qkv[p + d + 16] = (_Float16)(x1 * c + x0 * sn);
    }
}

// ---------------- flash attention (transposed tiles): one wave per (head, 16 queries) ----
// Sᵀ = K·Qᵀ so each lane's column is ONE query: softmax stats are in-lane
// (15 fmax/add) + a single shfl_xor(16); ctxᵀ = Vᵀ·Pᵀ keeps the rescale factor
// and 1/l lane-local (no broadcast shuffles at all).
__global__ __launch_bounds__(256) void attn_kernel(const _Float16* __restrict__ qkv,
                                                   _Float16* __restrict__ ctx) {
    __shared__ _Float16 Pl[8][16 * 32];             // per-wave P tile, [query][key]
    const int wid  = threadIdx.x >> 5;
    const int lane = threadIdx.x & 31;
    const int hh   = lane >> 4;
    const int l16  = lane & 15;
    const int task = blockIdx.x * 8 + wid;          // NH * (S/16) = 2048 tasks
    const int head = task >> 7;
    const int qb   = task & 127;
    const int q0   = qb * 16;
    const int RS   = H3_;
    const size_t qbase = (size_t)head * (3 * HD_);

    // ---- Q as B-operand tiles: B[k=d][n=query], pairs contiguous in Q rows ----
    v16h qB[4];
    {
        const _Float16* qrow = qkv + (size_t)(q0 + l16) * RS + qbase;
        #pragma unroll
        for (int kk = 0; kk < 4; ++kk) {
            Pack16 t;
            #pragma unroll
            for (int j = 0; j < 8; ++j)
                t.u[j] = *(const unsigned*)(qrow + kk * 32 + ((j >> 2) << 4) + ((j & 3) << 1) + hh * 8);
            qB[kk] = as_v16h(t);
        }
    }

    v8f o[8] = {};                                  // ctxᵀ: rows=d (8 tiles), col=own query
    float m = -1e30f, lsum = 0.f;                   // lane-local stats (query = l16)

    const float scale = 0.08838834764831845f;       // 1/sqrt(128)
    const int nkb = (q0 + 16 + 31) / 32;

    for (int it = 0; it < nkb; ++it) {
        const int kb = it * 32;
        const bool masked = (kb + 31 > q0);         // uniform branch; tail blocks only
        if (kb + 32 < S_)
            __builtin_prefetch(qkv + (size_t)(kb + 32 + l16) * RS + qbase + HD_, 0, 1);

        // ---- K as A-operand tiles: rows=keys, K-dim=d (2 key halves x 4 chunks) ----
        Pack16 kt8[8];
        #pragma unroll
        for (int half = 0; half < 2; ++half) {
            const _Float16* kr = qkv + (size_t)(kb + half * 16 + l16) * RS + qbase + HD_;
            #pragma unroll
            for (int kk = 0; kk < 4; ++kk) {
                #pragma unroll
                for (int v = 0; v < 8; ++v) {
                    int kp = ((v >> 2) << 4) + ((v & 3) << 1) + hh * 8;
                    kt8[half * 4 + kk].u[v] = *(const unsigned*)(kr + kk * 32 + kp);
                }
            }
        }
        // ---- Sᵀ tiles: rows=keys, cols=queries ----
        v8f sc[2] = {};
        #pragma unroll
        for (int half = 0; half < 2; ++half)
            #pragma unroll
            for (int kk = 0; kk < 4; ++kk)
                sc[half] = wmma_f16(as_v16h(kt8[half * 4 + kk]), qB[kk], sc[half]);

        // ---- per-query (lane-local) online softmax over 16 in-lane values ----
        float mloc = -1e30f;
        #pragma unroll
        for (int half = 0; half < 2; ++half) {
            #pragma unroll
            for (int r = 0; r < 8; ++r) {
                float v = sc[half][r] * scale;
                if (masked) {
                    int key = kb + half * 16 + r + 8 * hh;
                    if (key > q0 + l16) v = -1e30f;
                }
                sc[half][r] = v;
                mloc = fmaxf(mloc, v);
            }
        }
        mloc = fmaxf(mloc, __shfl_xor(mloc, 16, 32));
        float mn = fmaxf(m, mloc);
        float sum = 0.f;
        #pragma unroll
        for (int half = 0; half < 2; ++half) {
            #pragma unroll
            for (int r = 0; r < 8; ++r) {
                float p = expf(sc[half][r] - mn);
                sum += p;
                Pl[wid][l16 * 32 + half * 16 + r + 8 * hh] = (_Float16)p;
            }
        }
        sum += __shfl_xor(sum, 16, 32);
        float f = expf(m - mn);
        lsum = lsum * f + sum;
        m = mn;
        #pragma unroll
        for (int tt = 0; tt < 8; ++tt)
            #pragma unroll
            for (int r = 0; r < 8; ++r)
                o[tt][r] *= f;

        // ---- Vᵀ as A-operand tiles: rows=d (8 tiles of 16), K-dim=keys ----
        Pack16 vt8[8];
        #pragma unroll
        for (int tt = 0; tt < 8; ++tt) {
            #pragma unroll
            for (int v = 0; v < 8; ++v) {
                int kp = ((v >> 2) << 4) + ((v & 3) << 1) + hh * 8;
                size_t vb = (size_t)(kb + kp) * RS + qbase + 2 * HD_ + tt * 16 + l16;
                unsigned lo = __builtin_bit_cast(unsigned short, qkv[vb]);
                unsigned hi = __builtin_bit_cast(unsigned short, qkv[vb + RS]);
                vt8[tt].u[v] = lo | (hi << 16);
            }
        }

        asm volatile("s_wait_dscnt 0x0" ::: "memory");   // wave-local LDS visibility

        // ---- Pᵀ as B operand from LDS: B[k=key][n=query], pairs contiguous ----
        Pack16 tb;
        #pragma unroll
        for (int j = 0; j < 8; ++j)
            tb.u[j] = *(const unsigned*)&Pl[wid][l16 * 32 + ((j >> 2) << 4) + ((j & 3) << 1) + hh * 8];
        v16h pB = as_v16h(tb);

        // ---- ctxᵀ += Vᵀ(128x32) x Pᵀ(32x16), 8 d-tiles ----
        #pragma unroll
        for (int tt = 0; tt < 8; ++tt)
            o[tt] = wmma_f16(as_v16h(vt8[tt]), pB, o[tt]);
    }

    // ---- normalize (lane-local 1/l) and write ctx[query][head*HD + d] ----
    float inv = 1.0f / lsum;
    #pragma unroll
    for (int tt = 0; tt < 8; ++tt) {
        #pragma unroll
        for (int r = 0; r < 8; ++r) {
            int d = tt * 16 + r + 8 * hh;
            ctx[(size_t)(q0 + l16) * H_ + head * HD_ + d] = (_Float16)(o[tt][r] * inv);
        }
    }
}

// ---------------- generic WMMA GEMM: C[M,N] = A[M,K] @ W[N,K]^T + bias (+res) ----------------
// block = 256 threads (8 waves), C tile 128x128, wave tile 32x64, K-step 32,
// double-buffered LDS A tile filled via async global->LDS when available.
template<bool ACT_GELU, bool OUT_HALF, bool HAS_BIAS, bool HAS_RES>
__global__ __launch_bounds__(256) void gemm_wmma_kernel(
    const _Float16* __restrict__ A,
    const _Float16* __restrict__ W,
    const float* __restrict__ bias,
    const float* __restrict__ res1,
    const float* __restrict__ res2,
    void* __restrict__ out,
    int M, int N, int K)
{
    __shared__ _Float16 As[2][128 * 32];            // 2 x 8 KB A tiles
    const int n0 = blockIdx.x * 128;
    const int m0 = blockIdx.y * 128;
    const int tid  = threadIdx.x;
    const int wid  = tid >> 5;
    const int lane = tid & 31;
    const int wm = wid >> 1, wn = wid & 1;          // 4 x 2 waves, 32 x 64 per wave
    const int hh = lane >> 4, l16 = lane & 15;

    const int srow = tid >> 1, shalf = tid & 1;     // staging: 32B per thread

    v8f acc[2][4] = {};
    const int nkt = K / 32;

    // stage first tile
    {
        const _Float16* src = A + (size_t)(m0 + srow) * K + shalf * 16;
        copy16B_to_lds(src,     &As[0][srow * 32 + shalf * 16]);
        copy16B_to_lds(src + 8, &As[0][srow * 32 + shalf * 16 + 8]);
    }

    for (int kt = 0; kt < nkt; ++kt) {
        const int k0 = kt * 32;
        const int cur = kt & 1;
        if (kt + 1 < nkt) {                         // stage next tile into other buffer
            const _Float16* src = A + (size_t)(m0 + srow) * K + (k0 + 32) + shalf * 16;
            copy16B_to_lds(src,     &As[cur ^ 1][srow * 32 + shalf * 16]);
            copy16B_to_lds(src + 8, &As[cur ^ 1][srow * 32 + shalf * 16 + 8]);
            wait_async_prev();
            __builtin_prefetch(W + (size_t)(n0 + wn * 64 + l16) * K + k0 + 32, 0, 1);
        } else {
            wait_async_all();
        }
        __syncthreads();

        // ---- A operands from LDS (two 16-row subtiles) ----
        v16h a[2];
        #pragma unroll
        for (int sm = 0; sm < 2; ++sm) {
            Pack16 t;
            int row = wm * 32 + sm * 16 + l16;
            #pragma unroll
            for (int v = 0; v < 8; ++v) {
                int kp = ((v >> 2) << 4) + ((v & 3) << 1) + hh * 8;
                t.u[v] = *(const unsigned*)&As[cur][row * 32 + kp];
            }
            a[sm] = as_v16h(t);
        }
        // ---- 4 B operands straight from global (K contiguous in W rows) ----
        v16h b[4];
        #pragma unroll
        for (int sn = 0; sn < 4; ++sn) {
            Pack16 t;
            int col = n0 + wn * 64 + sn * 16 + l16;
            const _Float16* wp = W + (size_t)col * K + k0 + hh * 8;
            #pragma unroll
            for (int j = 0; j < 8; ++j)
                t.u[j] = *(const unsigned*)(wp + ((j >> 2) << 4) + ((j & 3) << 1));
            b[sn] = as_v16h(t);
        }
        #pragma unroll
        for (int sm = 0; sm < 2; ++sm)
            #pragma unroll
            for (int sn = 0; sn < 4; ++sn)
                acc[sm][sn] = wmma_f16(a[sm], b[sn], acc[sm][sn]);
        __syncthreads();
    }

    // ---- epilogue: bias / gelu / residual adds / store ----
    #pragma unroll
    for (int sm = 0; sm < 2; ++sm) {
        #pragma unroll
        for (int sn = 0; sn < 4; ++sn) {
            int col = n0 + wn * 64 + sn * 16 + l16;
            float bv = HAS_BIAS ? bias[col] : 0.0f;
            #pragma unroll
            for (int r = 0; r < 8; ++r) {
                int row = m0 + wm * 32 + sm * 16 + r + 8 * hh;
                float v = acc[sm][sn][r] + bv;
                if (ACT_GELU) {
                    float t = 0.79788456f * v * (1.0f + 0.044715f * v * v);
                    v = v * 0.5f * (1.0f + tanhf(t));
                }
                size_t idx = (size_t)row * N + col;
                if (HAS_RES) v += res1[idx] + res2[idx];
                if (OUT_HALF) ((_Float16*)out)[idx] = (_Float16)v;
                else          ((float*)out)[idx] = v;
            }
        }
    }
}

// ---------------- host-side orchestration ----------------
extern "C" void kernel_launch(void* const* d_in, const int* in_sizes, int n_in,
                              void* d_out, int out_size, void* d_ws, size_t ws_size,
                              hipStream_t stream) {
    (void)in_sizes; (void)n_in; (void)out_size; (void)ws_size;
    const int*   tokens     = (const int*)  d_in[0];
    const float* embed      = (const float*)d_in[1];
    const float* ln1_g      = (const float*)d_in[2];
    const float* ln1_b      = (const float*)d_in[3];
    const float* ln2_g      = (const float*)d_in[4];
    const float* ln2_b      = (const float*)d_in[5];
    const float* qkv_w      = (const float*)d_in[6];
    const float* qkv_b      = (const float*)d_in[7];
    const float* attn_out_w = (const float*)d_in[8];
    const float* attn_out_b = (const float*)d_in[9];
    const float* fc1_w      = (const float*)d_in[10];
    const float* fc1_b      = (const float*)d_in[11];
    const float* fc2_w      = (const float*)d_in[12];
    const float* fc2_b      = (const float*)d_in[13];
    const float* lnf_g      = (const float*)d_in[14];
    const float* lnf_b      = (const float*)d_in[15];
    const float* logits_w   = (const float*)d_in[16];

    // ---- workspace layout ----
    uint8_t* w = (uint8_t*)d_ws;
    float*    hA       = (float*)w;     w += (size_t)S_ * H_ * 4;
    float*    hB       = (float*)w;     w += (size_t)S_ * H_ * 4;
    float*    attn_out = (float*)w;     w += (size_t)S_ * H_ * 4;
    _Float16* xln      = (_Float16*)w;  w += (size_t)S_ * H_ * 2;
    _Float16* qkv      = (_Float16*)w;  w += (size_t)S_ * H3_ * 2;
    _Float16* ctx      = (_Float16*)w;  w += (size_t)S_ * H_ * 2;
    _Float16* act1     = (_Float16*)w;  w += (size_t)S_ * H4_ * 2;
    _Float16* qkvw_h   = (_Float16*)w;  w += (size_t)L_ * H3_ * H_ * 2;
    _Float16* attnw_h  = (_Float16*)w;  w += (size_t)L_ * H_ * H_ * 2;
    _Float16* fc1w_h   = (_Float16*)w;  w += (size_t)L_ * H4_ * H_ * 2;
    _Float16* fc2w_h   = (_Float16*)w;  w += (size_t)L_ * H_ * H4_ * 2;
    _Float16* logw_h   = (_Float16*)w;  w += (size_t)V_ * H_ * 2;

    // ---- 1) convert weights fp32 -> f16 once ----
    const int CVTB = 4096;
    cvt_f32_to_f16<<<CVTB, 256, 0, stream>>>(qkv_w,      qkvw_h,  (size_t)L_ * H3_ * H_);
    cvt_f32_to_f16<<<CVTB, 256, 0, stream>>>(attn_out_w, attnw_h, (size_t)L_ * H_ * H_);
    cvt_f32_to_f16<<<CVTB, 256, 0, stream>>>(fc1_w,      fc1w_h,  (size_t)L_ * H4_ * H_);
    cvt_f32_to_f16<<<CVTB, 256, 0, stream>>>(fc2_w,      fc2w_h,  (size_t)L_ * H_ * H4_);
    cvt_f32_to_f16<<<CVTB, 256, 0, stream>>>(logits_w,   logw_h,  (size_t)V_ * H_);

    // ---- 2) embedding ----
    embed_kernel<<<S_, 256, 0, stream>>>(tokens, embed, hA);

    float* hin = hA;
    float* hout = hB;
    for (int li = 0; li < L_; ++li) {
        // attn path
        layernorm_kernel<<<S_, 256, 0, stream>>>(hin, ln1_g + li * H_, ln1_b + li * H_, xln);
        gemm_wmma_kernel<false, true, true, false><<<dim3(H3_ / 128, S_ / 128), 256, 0, stream>>>(
            xln, qkvw_h + (size_t)li * H3_ * H_, qkv_b + li * H3_, nullptr, nullptr, qkv, S_, H3_, H_);
        rope_kernel<<<(S_ * NH_ * 16 + 255) / 256, 256, 0, stream>>>(qkv);
        attn_kernel<<<NH_ * (S_ / 16) / 8, 256, 0, stream>>>(qkv, ctx);
        gemm_wmma_kernel<false, false, true, false><<<dim3(H_ / 128, S_ / 128), 256, 0, stream>>>(
            ctx, attnw_h + (size_t)li * H_ * H_, attn_out_b + li * H_, nullptr, nullptr, attn_out, S_, H_, H_);
        // mlp path
        layernorm_kernel<<<S_, 256, 0, stream>>>(hin, ln2_g + li * H_, ln2_b + li * H_, xln);
        gemm_wmma_kernel<true, true, true, false><<<dim3(H4_ / 128, S_ / 128), 256, 0, stream>>>(
            xln, fc1w_h + (size_t)li * H4_ * H_, fc1_b + li * H4_, nullptr, nullptr, act1, S_, H4_, H_);
        // fc2 fused: h_new = h_old + attn_out + (act1 @ fc2^T + b)
        gemm_wmma_kernel<false, false, true, true><<<dim3(H_ / 128, S_ / 128), 256, 0, stream>>>(
            act1, fc2w_h + (size_t)li * H_ * H4_, fc2_b + li * H_, hin, attn_out, hout, S_, H_, H4_);
        float* tmp = hin; hin = hout; hout = tmp;
    }

    // ---- final LN + logits ----
    layernorm_kernel<<<S_, 256, 0, stream>>>(hin, lnf_g, lnf_b, xln);
    gemm_wmma_kernel<false, false, false, false><<<dim3(V_ / 128, S_ / 128), 256, 0, stream>>>(
        xln, logw_h, nullptr, nullptr, nullptr, (float*)d_out, S_, V_, H_);
}
